// FPLPGCN_dw_linear_1168231104597
// MI455X (gfx1250) — matrix operand order repaired and checked by reading, not compiled, not run on hardware
//
#include <hip/hip_runtime.h>
#include <hip/hip_bf16.h>

typedef __attribute__((ext_vector_type(16))) _Float16 v16h;
typedef __attribute__((ext_vector_type(8)))  float    v8f;

#define TPB 256

// ---------------------------------------------------------------- utility
__global__ __launch_bounds__(TPB) void k_fill(float* __restrict__ p, float v, long n) {
  long i = (long)blockIdx.x * blockDim.x + threadIdx.x;
  if (i < n) p[i] = v;
}

__global__ __launch_bounds__(TPB) void k_deg_edges(const int* __restrict__ dst,
                                                   float* __restrict__ deg, long E) {
  long e = (long)blockIdx.x * blockDim.x + threadIdx.x;
  if (e < E) atomicAdd(&deg[dst[e]], 1.0f);
}

__global__ __launch_bounds__(TPB) void k_rsqrt_inplace(float* __restrict__ p, long n) {
  long i = (long)blockIdx.x * blockDim.x + threadIdx.x;
  if (i < n) p[i] = rsqrtf(p[i]);
}

__global__ __launch_bounds__(TPB) void k_norm(const int* __restrict__ src,
                                              const int* __restrict__ dst,
                                              const float* __restrict__ dinv,
                                              float* __restrict__ nrm, long E) {
  long e = (long)blockIdx.x * blockDim.x + threadIdx.x;
  if (e < E) nrm[e] = dinv[src[e]] * dinv[dst[e]];
}

// -------------------------------------------------- operand packing (f32 -> f16 WMMA layout)
// A layout (16-bit A, 16x32 per tile-step), ISA 7.12.2:
//   lane = khalf*16 + m  (m = M row 0..15, khalf = 0/1)
//   half index i (0..15): k = ks*32 + khalf*8 + (i<8 ? i : i+8)
__global__ __launch_bounds__(TPB) void k_pack_a(const float* __restrict__ X,
                                                _Float16* __restrict__ pA,
                                                int M, int K) {
  long idx = (long)blockIdx.x * blockDim.x + threadIdx.x;
  if (idx >= (long)M * K) return;
  int  i    = (int)(idx & 15);
  int  lane = (int)((idx >> 4) & 31);
  long t    = idx >> 9;                 // tileM*Ksteps + ks
  int  Ks   = K >> 5;
  int  tileM = (int)(t / Ks);
  int  ks    = (int)(t % Ks);
  int  m     = lane & 15;
  int  khalf = lane >> 4;
  int  k = ks * 32 + khalf * 8 + ((i < 8) ? i : (i + 8));
  pA[idx] = (_Float16)X[(long)(tileM * 16 + m) * K + k];
}

// Fused-feature pack: logical A row = concat(h[64], xl[32], dw[64]), K = 160.
__global__ __launch_bounds__(TPB) void k_pack_a_fused(const float* __restrict__ H,
                                                      const float* __restrict__ XL,
                                                      const float* __restrict__ DWE,
                                                      _Float16* __restrict__ pA, int M) {
  const int K = 160;
  long idx = (long)blockIdx.x * blockDim.x + threadIdx.x;
  if (idx >= (long)M * K) return;
  int  i    = (int)(idx & 15);
  int  lane = (int)((idx >> 4) & 31);
  long t    = idx >> 9;
  int  Ks   = K >> 5;                   // 5
  int  tileM = (int)(t / Ks);
  int  ks    = (int)(t % Ks);
  int  m     = lane & 15;
  int  khalf = lane >> 4;
  int  k = ks * 32 + khalf * 8 + ((i < 8) ? i : (i + 8));
  long row = (long)tileM * 16 + m;
  float v;
  if      (k < 64) v = H  [row * 64 + k];
  else if (k < 96) v = XL [row * 32 + (k - 64)];
  else             v = DWE[row * 64 + (k - 96)];
  pA[idx] = (_Float16)v;
}

// B layout (16-bit B, 32x16 per tile-step): lane = nhalf*16 + n,
//   half i (0..15): k = ks*32 + (lane>>4)*8 + (i<8 ? i : i+8) ; col = tileN*16 + n
__global__ __launch_bounds__(TPB) void k_pack_b(const float* __restrict__ W,
                                                _Float16* __restrict__ pB,
                                                int K, int F) {
  long idx = (long)blockIdx.x * blockDim.x + threadIdx.x;
  if (idx >= (long)K * F) return;
  int  i    = (int)(idx & 15);
  int  lane = (int)((idx >> 4) & 31);
  long t    = idx >> 9;                 // tileN*Ksteps + ks
  int  Ks   = K >> 5;
  int  tileN = (int)(t / Ks);
  int  ks    = (int)(t % Ks);
  int  n  = lane & 15;
  int  k  = ks * 32 + (lane >> 4) * 8 + ((i < 8) ? i : (i + 8));
  pB[idx] = (_Float16)W[(long)k * F + tileN * 16 + n];
}

// ---------------------------------------------------------------- WMMA GEMM
// Block = 8 waves. The whole B panel (NT*KS KB) is staged once per block into
// LDS with global_load_async_to_lds_b128 (ASYNCcnt), then each wave owns one
// 16-row strip: A fragments live in registers across all NT column tiles and
// B comes from LDS (ds_load_b128). F == NT*16.
template <int KS, int NT>
__global__ __launch_bounds__(TPB) void k_gemm(const _Float16* __restrict__ pA,
                                              const _Float16* __restrict__ pB,
                                              float* __restrict__ out,
                                              int MT) {
  __shared__ _Float16 ldsB[NT * KS * 512];
  const int tid  = threadIdx.x;
  const int lane = tid & 31;
  const int wv   = tid >> 5;

  // ---- async-stage B panel into LDS (16B per lane per op, GV addressing) ----
  {
    const int CH = NT * KS * 64;                 // # of 16-byte chunks
    unsigned lbase = (unsigned)(size_t)(void*)ldsB;  // LDS aperture low bits = LDS offset
    for (int c = tid; c < CH; c += TPB) {
      unsigned    loff = lbase + (unsigned)c * 16u;
      const char* g    = (const char*)pB + (size_t)c * 16u;
      asm volatile("global_load_async_to_lds_b128 %0, %1, off"
                   :: "v"(loff), "v"(g) : "memory");
    }
    asm volatile("s_wait_asynccnt 0x0" ::: "memory");
  }
  __syncthreads();

  int tileM = blockIdx.x * (TPB / 32) + wv;
  if (tileM >= MT) return;                       // wave-uniform; EXEC stays all-ones

  // A strip for this wave, held in registers across all column tiles
  const v16h* ap = (const v16h*)pA + (size_t)tileM * KS * 32 + lane;
  v16h a[KS];
#pragma unroll
  for (int ks = 0; ks < KS; ++ks) a[ks] = ap[ks * 32];

  const v16h* bp = (const v16h*)ldsB + lane;
  long rowBase = (long)tileM * 16 + ((lane >> 4) * 8);
  int  colBase = lane & 15;
#pragma unroll
  for (int nt = 0; nt < NT; ++nt) {
    v8f c = {};
#pragma unroll
    for (int ks = 0; ks < KS; ++ks) {
      v16h b = bp[(nt * KS + ks) * 32];
      c = __builtin_amdgcn_wmma_f32_16x16x32_f16(false, a[ks], false, b,
                                                 (short)0, c, false, false);
    }
    // C/D layout: VGPR r -> M = (lane>=16 ? 8 : 0) + r, N = lane&15
    int col = nt * 16 + colBase;
#pragma unroll
    for (int r = 0; r < 8; ++r)
      out[(rowBase + r) * (long)(NT * 16) + col] = c[r];
  }
}

// ---------------------------------------------------------------- aggregation
// out = inv_deg * hW + bias   (self-loop term; also initializes accumulator)
__global__ __launch_bounds__(TPB) void k_agg_init(const float* __restrict__ hW,
                                                  const float* __restrict__ dinv,
                                                  const float* __restrict__ bias,
                                                  float* __restrict__ out,
                                                  long n, int F) {
  long idx = (long)blockIdx.x * blockDim.x + threadIdx.x;
  if (idx >= n) return;
  long i = idx / F;
  int  f = (int)(idx % F);
  float di = dinv[i];
  out[idx] = di * di * hW[idx] + bias[f];
}

// out[dst] += norm[e] * hW[src], float4 chunks; all node arrays are L2-resident
__global__ __launch_bounds__(TPB) void k_agg_edges(const int* __restrict__ src,
                                                   const int* __restrict__ dst,
                                                   const float* __restrict__ nrm,
                                                   const float* __restrict__ hW,
                                                   float* __restrict__ out,
                                                   long nwork, int F) {
  long idx = (long)blockIdx.x * blockDim.x + threadIdx.x;
  if (idx >= nwork) return;
  int  ch = F >> 2;
  long e  = idx / ch;
  int  c  = (int)(idx % ch);
  int   s = src[e];
  int   d = dst[e];
  float w = nrm[e];
  const float4 v = *(const float4*)(hW + (long)s * F + c * 4);
  float* o = out + (long)d * F + c * 4;
  atomicAdd(o + 0, w * v.x);
  atomicAdd(o + 1, w * v.y);
  atomicAdd(o + 2, w * v.z);
  atomicAdd(o + 3, w * v.w);
}

__global__ __launch_bounds__(TPB) void k_mask_merge(const int* __restrict__ mask,
                                                    const float* __restrict__ y,
                                                    const float* __restrict__ agg,
                                                    float* __restrict__ xl,
                                                    long n, int F) {
  long idx = (long)blockIdx.x * blockDim.x + threadIdx.x;
  if (idx >= n) return;
  long i = idx / F;
  xl[idx] = mask[i] ? y[idx] : agg[idx];
}

__global__ __launch_bounds__(TPB) void k_bias_sigmoid(float* __restrict__ out,
                                                      const float* __restrict__ b,
                                                      long n, int F) {
  long idx = (long)blockIdx.x * blockDim.x + threadIdx.x;
  if (idx >= n) return;
  float v = out[idx] + b[idx % F];
  out[idx] = 1.0f / (1.0f + expf(-v));
}

// ---------------------------------------------------------------- host driver
extern "C" void kernel_launch(void* const* d_in, const int* in_sizes, int n_in,
                              void* d_out, int out_size, void* d_ws, size_t ws_size,
                              hipStream_t stream) {
  const int IN_DIM = 128, HID = 64, OUTF = 32, DWF = 64, NLBL = 10;
  const int KF = HID + OUTF + DWF;  // 160

  const float* x    = (const float*)d_in[0];
  const float* y    = (const float*)d_in[1];
  const int*   ei   = (const int*)  d_in[2];
  const float* dwe  = (const float*)d_in[3];
  const int*   mask = (const int*)  d_in[4];
  const float* Wg0  = (const float*)d_in[5];
  const float* bg0  = (const float*)d_in[6];
  const float* Wg1  = (const float*)d_in[7];
  const float* bg1  = (const float*)d_in[8];
  const float* Wl   = (const float*)d_in[9];   // [10][32][32]
  const float* bl   = (const float*)d_in[10];  // [10][32]
  const float* Wf   = (const float*)d_in[11];  // [160][32]
  const float* bf   = (const float*)d_in[12];
  float* out = (float*)d_out;

  const long Nn = in_sizes[0] / IN_DIM;
  const long Ee = in_sizes[2] / 2;
  const int  MT = (int)(Nn >> 4);
  const int* src = ei;
  const int* dst = ei + Ee;

  // workspace carve-up (256B aligned)
  char* w = (char*)d_ws;
  auto carve = [&](size_t bytes) -> char* {
    char* p = w;
    w += (bytes + 255) & ~(size_t)255;
    return p;
  };
  float*     dinv   = (float*)    carve((size_t)Nn * 4);
  float*     nrm    = (float*)    carve((size_t)Ee * 4);
  _Float16*  bufA   = (_Float16*) carve((size_t)Nn * KF * 2);
  float*     bufHW  = (float*)    carve((size_t)Nn * HID * 4);
  float*     bufH   = (float*)    carve((size_t)Nn * HID * 4);
  float*     bufXL  = (float*)    carve((size_t)Nn * OUTF * 4);
  float*     bufXL2 = (float*)    carve((size_t)Nn * OUTF * 4);
  _Float16*  pW0    = (_Float16*) carve((size_t)IN_DIM * HID * 2);
  _Float16*  pW1    = (_Float16*) carve((size_t)HID * HID * 2);
  _Float16*  pWl    = (_Float16*) carve((size_t)NLBL * OUTF * OUTF * 2);
  _Float16*  pWf    = (_Float16*) carve((size_t)KF * OUTF * 2);

  auto nb = [](long n) { return (unsigned)((n + TPB - 1) / TPB); };
  const unsigned gemmBlocks = (unsigned)((MT + 7) / 8);   // 8 waves per block

  // ---- pack weights (tiny) ----
  k_pack_b<<<nb((long)IN_DIM * HID), TPB, 0, stream>>>(Wg0, pW0, IN_DIM, HID);
  k_pack_b<<<nb((long)HID * HID),    TPB, 0, stream>>>(Wg1, pW1, HID, HID);
  for (int j = 0; j < NLBL; ++j)
    k_pack_b<<<nb((long)OUTF * OUTF), TPB, 0, stream>>>(Wl + (long)j * OUTF * OUTF,
                                                        pWl + (long)j * OUTF * OUTF,
                                                        OUTF, OUTF);
  k_pack_b<<<nb((long)KF * OUTF), TPB, 0, stream>>>(Wf, pWf, KF, OUTF);

  // ---- GCN normalization: deg = indeg+1, dinv = rsqrt(deg), norm[e] ----
  k_fill<<<nb(Nn), TPB, 0, stream>>>(dinv, 1.0f, Nn);
  k_deg_edges<<<nb(Ee), TPB, 0, stream>>>(dst, dinv, Ee);
  k_rsqrt_inplace<<<nb(Nn), TPB, 0, stream>>>(dinv, Nn);
  k_norm<<<nb(Ee), TPB, 0, stream>>>(src, dst, dinv, nrm, Ee);

  // ---- GCN layer 0: x(128) -> bufH(64) ----
  k_pack_a<<<nb(Nn * IN_DIM), TPB, 0, stream>>>(x, bufA, (int)Nn, IN_DIM);
  k_gemm<4, 4><<<gemmBlocks, TPB, 0, stream>>>(bufA, pW0, bufHW, MT);
  k_agg_init<<<nb(Nn * HID), TPB, 0, stream>>>(bufHW, dinv, bg0, bufH, Nn * HID, HID);
  k_agg_edges<<<nb(Ee * (HID / 4)), TPB, 0, stream>>>(src, dst, nrm, bufHW, bufH,
                                                      Ee * (HID / 4), HID);

  // ---- GCN layer 1: bufH(64) -> bufH(64) (safe: pack_a consumes old bufH first)
  k_pack_a<<<nb(Nn * HID), TPB, 0, stream>>>(bufH, bufA, (int)Nn, HID);
  k_gemm<2, 4><<<gemmBlocks, TPB, 0, stream>>>(bufA, pW1, bufHW, MT);
  k_agg_init<<<nb(Nn * HID), TPB, 0, stream>>>(bufHW, dinv, bg1, bufH, Nn * HID, HID);
  k_agg_edges<<<nb(Ee * (HID / 4)), TPB, 0, stream>>>(src, dst, nrm, bufHW, bufH,
                                                      Ee * (HID / 4), HID);

  // ---- label propagation: xl = y; 10x (conv then masked overwrite) ----
  const float* xin = y;
  for (int j = 0; j < NLBL; ++j) {
    k_pack_a<<<nb(Nn * OUTF), TPB, 0, stream>>>(xin, bufA, (int)Nn, OUTF);
    k_gemm<1, 2><<<gemmBlocks, TPB, 0, stream>>>(bufA, pWl + (long)j * OUTF * OUTF,
                                                 bufHW, MT);
    k_agg_init<<<nb(Nn * OUTF), TPB, 0, stream>>>(bufHW, dinv, bl + (long)j * OUTF,
                                                  bufXL2, Nn * OUTF, OUTF);
    k_agg_edges<<<nb(Ee * (OUTF / 4)), TPB, 0, stream>>>(src, dst, nrm, bufHW, bufXL2,
                                                         Ee * (OUTF / 4), OUTF);
    k_mask_merge<<<nb(Nn * OUTF), TPB, 0, stream>>>(mask, y, bufXL2, bufXL,
                                                    Nn * OUTF, OUTF);
    xin = bufXL;
  }

  // ---- fuse: concat(h, xl, dw)(160) @ W_fuse(160x32) + b, sigmoid ----
  k_pack_a_fused<<<nb(Nn * KF), TPB, 0, stream>>>(bufH, bufXL, dwe, bufA, (int)Nn);
  k_gemm<5, 2><<<gemmBlocks, TPB, 0, stream>>>(bufA, pWf, out, MT);
  k_bias_sigmoid<<<nb(Nn * OUTF), TPB, 0, stream>>>(out, bf, Nn * OUTF, OUTF);
}
